// VectorQuantizedVariationalAutoEncoder_4398046511635
// MI455X (gfx1250) — matrix operand order repaired and checked by reading, not compiled, not run on hardware
//
#include <hip/hip_runtime.h>
#include <hip/hip_bf16.h>

// ---------------------------------------------------------------------------
// VQ-VAE vector quantization for MI455X (gfx1250, wave32, WMMA).
//   latent [32,256,32,32] f32, codebook [2048,256] f32
//   out    = quantized [32,256,32,32] f32  ++  vq_loss scalar f32
// Score GEMM (32768x2048x256) on v_wmma_f32_16x16x32_bf16, B panels staged
// through LDS with global_load_async_to_lds_b128 (ASYNCcnt double-buffer).
// ---------------------------------------------------------------------------

#define C_DIM 256
#define K_CB  2048
#define HW    1024
#define B_SZ  32
#define NROWS (B_SZ * HW)              // 32768 latent vectors
#define OUT_ELEMS (B_SZ * C_DIM * HW)  // 8388608
#define PANEL_BYTES (16 * C_DIM * 2)   // 16 codewords x 256 bf16 = 8192 B
#define NTILES (K_CB / 16)             // 128 column tiles

typedef __attribute__((ext_vector_type(16))) __bf16 v16bf;
typedef __attribute__((ext_vector_type(8)))  float  v8f;

union V16 { v16bf v; uint4 q[2]; };

// ---------------------------------------------------------------------------
// 1) Codebook prep: f32 -> bf16 staging + row squared norms
// ---------------------------------------------------------------------------
__global__ void __launch_bounds__(256)
vq_prep_emb(const float* __restrict__ emb,
            __hip_bfloat16* __restrict__ ebf,
            float* __restrict__ e2) {
  const int k = blockIdx.x;
  const int c = threadIdx.x;
  const float v = emb[k * C_DIM + c];
  ebf[k * C_DIM + c] = __float2bfloat16(v);

  __shared__ float red[256];
  red[c] = v * v;
  __syncthreads();
  #pragma unroll
  for (int s = 128; s > 0; s >>= 1) {
    if (c < s) red[c] += red[c + s];
    __syncthreads();
  }
  if (c == 0) e2[k] = red[0];
}

// ---------------------------------------------------------------------------
// 2) Latent transpose+convert: [B,C,HW] f32 -> row-major [B*HW, C] bf16
// ---------------------------------------------------------------------------
__global__ void __launch_bounds__(256)
vq_transpose_x(const float* __restrict__ latent,
               __hip_bfloat16* __restrict__ xbf) {
  __shared__ float tile[32][33];
  const int tx = threadIdx.x;         // 0..31
  const int ty = threadIdx.y;         // 0..7
  const int n0 = blockIdx.x * 32;
  const int c0 = blockIdx.y * 32;
  const int b  = blockIdx.z;

  const float* src = latent + ((size_t)b * C_DIM) * HW;
  #pragma unroll
  for (int i = 0; i < 4; ++i) {
    int c = c0 + ty + i * 8;
    tile[ty + i * 8][tx] = src[(size_t)c * HW + n0 + tx];   // coalesced in n
  }
  __syncthreads();
  __hip_bfloat16* dst = xbf + ((size_t)b * HW) * C_DIM;
  #pragma unroll
  for (int i = 0; i < 4; ++i) {
    int n = n0 + ty + i * 8;
    dst[(size_t)n * C_DIM + c0 + tx] = __float2bfloat16(tile[tx][ty + i * 8]); // coalesced in c
  }
}

// ---------------------------------------------------------------------------
// 3) WMMA argmin. 8 waves/block, 16 rows/wave. B panels (8 KB) staged into a
//    double-buffered LDS pipeline via global_load_async_to_lds_b128; all
//    waves consume them with ds_load_b128.
// ---------------------------------------------------------------------------
__global__ void __launch_bounds__(256)
vq_argmin(const __hip_bfloat16* __restrict__ xbf,
          const __hip_bfloat16* __restrict__ ebf,
          const float* __restrict__ e2,
          int* __restrict__ idx_out) {
  __shared__ __align__(16) unsigned char bpanel[2][PANEL_BYTES];

  const int tid  = threadIdx.x;
  const int lane = tid & 31;
  const int wave = (blockIdx.x * blockDim.x + tid) >> 5;
  const int row_base = wave * 16;          // 2048 waves * 16 rows = 32768
  const int half = lane >> 4;              // K-half select (A/B layouts)
  const int mrow = lane & 15;              // A: row within tile, B: column

  const unsigned long long ebase = (unsigned long long)(const void*)ebf;
  // Per-thread slice of the 8 KB panel: 2 x 16 B at +0 / +4096.
  const unsigned slice = (unsigned)(tid * 16);
  const unsigned lds0  = (unsigned)(size_t)&bpanel[0][0];  // low 32b == LDS offset
  const unsigned lds1  = (unsigned)(size_t)&bpanel[1][0];

  // --- Preload A: 16 rows x 256 bf16 pinned in 64 VGPRs.
  uint4 aq[8][2];
  const __hip_bfloat16* arow = xbf + (size_t)(row_base + mrow) * C_DIM;
  #pragma unroll
  for (int kk = 0; kk < 8; ++kk) {
    const int cb = kk * 32;
    aq[kk][0] = *(const uint4*)(arow + cb + half * 8);
    aq[kk][1] = *(const uint4*)(arow + cb + 16 + half * 8);
  }

  float bestv[8];
  int   besti[8];
  #pragma unroll
  for (int r = 0; r < 8; ++r) { bestv[r] = 3.4e38f; besti[r] = 0; }

  // --- Prologue: stage panel 0 into buffer 0.
  {
    unsigned g0 = slice;
    unsigned l0 = lds0 + slice;
    asm volatile("global_load_async_to_lds_b128 %0, %1, %2"
                 :: "v"(l0), "v"(g0), "s"(ebase) : "memory");
    asm volatile("global_load_async_to_lds_b128 %0, %1, %2"
                 :: "v"(l0 + 4096u), "v"(g0 + 4096u), "s"(ebase) : "memory");
  }
  asm volatile("s_wait_asynccnt 0x0" ::: "memory");
  __syncthreads();

  for (int ct = 0; ct < NTILES; ++ct) {
    const unsigned curbase = (ct & 1) ? lds1 : lds0;

    // Stage next panel into the other buffer (safe: last barrier proved all
    // waves are done reading it).
    if (ct + 1 < NTILES) {
      const unsigned nxtbase = (ct & 1) ? lds0 : lds1;
      const unsigned g0 = (unsigned)((ct + 1) * PANEL_BYTES) + slice;
      const unsigned l0 = nxtbase + slice;
      asm volatile("global_load_async_to_lds_b128 %0, %1, %2"
                   :: "v"(l0), "v"(g0), "s"(ebase) : "memory");
      asm volatile("global_load_async_to_lds_b128 %0, %1, %2"
                   :: "v"(l0 + 4096u), "v"(g0 + 4096u), "s"(ebase) : "memory");
    }

    const int col = ct * 16 + mrow;
    const float e2v = e2[col];
    const unsigned char* bp = &bpanel[ct & 1][0];

    v8f acc = {};
    #pragma unroll
    for (int kk = 0; kk < 8; ++kk) {
      const int cb = kk * 32;
      V16 a, bb;
      a.q[0] = aq[kk][0];
      a.q[1] = aq[kk][1];
      // B 32x16: lanes 0-15 hold K = cb+0..7,16..23, lanes 16-31 K = cb+8..15,24..31.
      const unsigned boff = (unsigned)(mrow * 512 + cb * 2 + half * 16);
      bb.q[0] = *(const uint4*)(bp + boff);        // ds_load_b128
      bb.q[1] = *(const uint4*)(bp + boff + 32);
      acc = __builtin_amdgcn_wmma_f32_16x16x32_bf16(
                false, a.v, false, bb.v, (short)0, acc, false, false);
    }

    // C/D layout: reg r -> row (r + 8*half), col = mrow (this lane's codeword)
    #pragma unroll
    for (int r = 0; r < 8; ++r) {
      const float cand = e2v - 2.0f * acc[r];
      if (cand < bestv[r]) { bestv[r] = cand; besti[r] = col; }
    }

    if (ct + 1 < NTILES)
      asm volatile("s_wait_asynccnt 0x0" ::: "memory");
    __syncthreads();
  }

  // Reduce across the 16 lanes of each half (masks 1..8 never cross bit 4).
  #pragma unroll
  for (int m = 1; m <= 8; m <<= 1) {
    #pragma unroll
    for (int r = 0; r < 8; ++r) {
      const float ov = __shfl_xor(bestv[r], m, 32);
      const int   oi = __shfl_xor(besti[r], m, 32);
      if (ov < bestv[r] || (ov == bestv[r] && oi < besti[r])) {
        bestv[r] = ov; besti[r] = oi;
      }
    }
  }
  if (mrow == 0) {
    #pragma unroll
    for (int r = 0; r < 8; ++r)
      idx_out[row_base + half * 8 + r] = besti[r];
  }
}

// ---------------------------------------------------------------------------
// 4) Gather winners, write transposed output coalesced in n, partial MSE.
// ---------------------------------------------------------------------------
__global__ void __launch_bounds__(256)
vq_gather(const float* __restrict__ latent,
          const float* __restrict__ emb,
          const int* __restrict__ idx,
          float* __restrict__ out,
          float* __restrict__ partial) {
  const int chunk = blockIdx.x;            // 128 = 32 batches * 4 n-chunks
  const int b = chunk >> 2;
  const int n = (chunk & 3) * 256 + threadIdx.x;
  const int k = idx[b * HW + n];

  const float* lb = latent + ((size_t)b * C_DIM) * HW;
  float*       ob = out    + ((size_t)b * C_DIM) * HW;
  const float* er = emb    + (size_t)k * C_DIM;

  float acc = 0.0f;
  for (int c = 0; c < C_DIM; ++c) {
    const float e = er[c];                 // codebook row: L2/WGP$ resident
    ob[(size_t)c * HW + n] = e;            // coalesced in n
    const float d = lb[(size_t)c * HW + n] - e;
    acc += d * d;
  }

  __shared__ float red[256];
  red[threadIdx.x] = acc;
  __syncthreads();
  #pragma unroll
  for (int s = 128; s > 0; s >>= 1) {
    if (threadIdx.x < s) red[threadIdx.x] += red[threadIdx.x + s];
    __syncthreads();
  }
  if (threadIdx.x == 0) partial[chunk] = red[0];
}

// ---------------------------------------------------------------------------
// 5) Final loss reduction (deterministic, single block).
// ---------------------------------------------------------------------------
__global__ void __launch_bounds__(128)
vq_loss_reduce(const float* __restrict__ partial, float* __restrict__ out) {
  __shared__ float red[128];
  red[threadIdx.x] = partial[threadIdx.x];
  __syncthreads();
  #pragma unroll
  for (int s = 64; s > 0; s >>= 1) {
    if (threadIdx.x < s) red[threadIdx.x] += red[threadIdx.x + s];
    __syncthreads();
  }
  if (threadIdx.x == 0)
    out[OUT_ELEMS] = red[0] * (1.0f / (float)OUT_ELEMS);
}

// ---------------------------------------------------------------------------
extern "C" void kernel_launch(void* const* d_in, const int* in_sizes, int n_in,
                              void* d_out, int out_size, void* d_ws, size_t ws_size,
                              hipStream_t stream) {
  const float* latent = (const float*)d_in[0];   // 32*256*32*32
  const float* emb    = (const float*)d_in[1];   // 2048*256
  float* out = (float*)d_out;                    // 8388608 quant + 1 loss

  // Workspace carve-up (~18 MB)
  char* ws = (char*)d_ws;
  __hip_bfloat16* xbf = (__hip_bfloat16*)ws;                       // 16.0 MiB
  ws += (size_t)NROWS * C_DIM * sizeof(__hip_bfloat16);
  __hip_bfloat16* ebf = (__hip_bfloat16*)ws;                       // 1.0 MiB
  ws += (size_t)K_CB * C_DIM * sizeof(__hip_bfloat16);
  float* e2 = (float*)ws;                                          // 8 KiB
  ws += (size_t)K_CB * sizeof(float);
  int* idx = (int*)ws;                                             // 128 KiB
  ws += (size_t)NROWS * sizeof(int);
  float* partial = (float*)ws;                                     // 512 B

  vq_prep_emb<<<K_CB, 256, 0, stream>>>(emb, ebf, e2);

  dim3 tgrid(HW / 32, C_DIM / 32, B_SZ);
  dim3 tblk(32, 8);
  vq_transpose_x<<<tgrid, tblk, 0, stream>>>(latent, xbf);

  // 32768 rows / (16 rows/wave * 8 waves/block) = 256 blocks
  vq_argmin<<<NROWS / (16 * 8), 256, 0, stream>>>(xbf, ebf, e2, idx);

  vq_gather<<<128, 256, 0, stream>>>(latent, emb, idx, out, partial);

  vq_loss_reduce<<<1, 128, 0, stream>>>(partial, out);
}
